// AttentionGraphLayer_1992864826008
// MI455X (gfx1250) — matrix operand order repaired and checked by reading, not compiled, not run on hardware
//
#include <hip/hip_runtime.h>
#include <hip/hip_bf16.h>
#include <cstdint>

typedef _Float16 v16h __attribute__((ext_vector_type(16)));
typedef _Float16 h2   __attribute__((ext_vector_type(2)));
typedef float    v8f  __attribute__((ext_vector_type(8)));
typedef int      v4i_t __attribute__((vector_size(4 * sizeof(int))));

#define B_  4
#define G_  8
#define N_  1024
#define D_  256
#define BG_ 32

#define TEMP_INV 2.0f      // 1 / 0.5
#define LN_EPS   1e-6f
#define EPS_SQ   1e-12f

#if __has_builtin(__builtin_amdgcn_global_load_async_to_lds_b128) && \
    __has_builtin(__builtin_amdgcn_s_wait_asynccnt)
#define USE_ASYNC_LDS 1
#else
#define USE_ASYNC_LDS 0
#endif

#define AS1 __attribute__((address_space(1)))
#define AS3 __attribute__((address_space(3)))

// ---- WMMA fragment index helpers (per CDNA5 ISA 16-bit layouts) ----
// A-matrix 16x32 (MxK): lane l -> M = l&15; element pair v (0..7):
//   K0 = 16*(v/4) + 2*(v%4) + 8*(l>>4)
__device__ __forceinline__ int a_k0(int vv, int half) {
  return ((vv >> 2) << 4) + ((vv & 3) << 1) + (half << 3);
}
// B-matrix 32x16 (KxN): lane l -> N = l&15; element pair v: K0 = 16*(v/4) + 2*(v%4) + 8*(l>>4)
__device__ __forceinline__ int b_k0(int vv, int half) {
  return ((vv >> 2) << 4) + ((vv & 3) << 1) + (half << 3);
}
// C/D 16x16 f32: lane l, reg r -> M = r + 8*(l>>4), N = l&15

__device__ __forceinline__ v8f wmma_f16(v16h a, v16h b, v8f c) {
  return __builtin_amdgcn_wmma_f32_16x16x32_f16(false, a, false, b, (short)0, c,
                                                false, false);
}

// ------------------------------------------------------------------
// Kernel 0: WT[e*D+k] = (f16) W[k*D+e]
// ------------------------------------------------------------------
__global__ void k0_transpose_w(const float* __restrict__ W,
                               _Float16* __restrict__ WT) {
  int tid = blockIdx.x * blockDim.x + threadIdx.x;
  int k = tid % D_;
  int e = tid / D_;
  WT[e * D_ + k] = (_Float16)W[k * D_ + e];
}

// ------------------------------------------------------------------
// Kernel 1: h = x@W + b  (f16 out) and kn = sum(h*h, d)
// grid (BG, N/64), block 128 (4 waves); each wave: 16 rows x 256 cols,
// done as two (unrolled) passes of 8 column-tiles.
// ------------------------------------------------------------------
__global__ void k1_gemm_h(const float* __restrict__ x,
                          const _Float16* __restrict__ WT,
                          const float* __restrict__ bias,
                          _Float16* __restrict__ h16,
                          float* __restrict__ kn) {
  const int lane = threadIdx.x & 31;
  const int wave = threadIdx.x >> 5;
  const int half = lane >> 4;
  const int lm   = lane & 15;
  const int bg   = blockIdx.x;
  const int row0 = blockIdx.y * 64 + wave * 16;

  const float* xg = x + (size_t)bg * N_ * D_;
  const float* xrow = xg + (size_t)(row0 + lm) * D_;   // A row for this lane
  const _Float16* wb0 = WT + (size_t)lm * D_;          // B base for this lane
  _Float16* hout = h16 + (size_t)bg * N_ * D_;

  const v8f vzero = {0.f, 0.f, 0.f, 0.f, 0.f, 0.f, 0.f, 0.f};
  float ss[8];
#pragma unroll
  for (int r = 0; r < 8; ++r) ss[r] = 0.f;

#pragma unroll
  for (int pass = 0; pass < 2; ++pass) {
    const int ct0 = pass * 8;
    v8f acc[8];
#pragma unroll
    for (int i = 0; i < 8; ++i) acc[i] = vzero;

    for (int kt = 0; kt < 8; ++kt) {
      const int kbase = kt * 32;
      // A fragment (f32 -> f16)
      v16h a;
#pragma unroll
      for (int vv = 0; vv < 8; ++vv) {
        const float2 f = *(const float2*)(xrow + kbase + a_k0(vv, half));
        a[2 * vv]     = (_Float16)f.x;
        a[2 * vv + 1] = (_Float16)f.y;
      }
      // B fragments: single base + compile-time offsets, one load burst
      v16h bf[8];
#pragma unroll
      for (int ct = 0; ct < 8; ++ct) {
#pragma unroll
        for (int vv = 0; vv < 8; ++vv) {
          const h2 t = *(const h2*)(wb0 + (size_t)(ct0 + ct) * 16 * D_ +
                                    kbase + b_k0(vv, half));
          bf[ct][2 * vv]     = t.x;
          bf[ct][2 * vv + 1] = t.y;
        }
      }
      // WMMA burst
#pragma unroll
      for (int ct = 0; ct < 8; ++ct) acc[ct] = wmma_f16(a, bf[ct], acc[ct]);
    }

    // epilogue for this half of the columns: bias, store h16, accumulate ss
#pragma unroll
    for (int ct = 0; ct < 8; ++ct) {
      const int col = (ct0 + ct) * 16 + lm;
      const float bb = bias[col];
#pragma unroll
      for (int r = 0; r < 8; ++r) {
        const float v = acc[ct][r] + bb;
        const int row = row0 + r + 8 * half;
        hout[(size_t)row * D_ + col] = (_Float16)v;
        ss[r] += v * v;
      }
    }
  }

#pragma unroll
  for (int r = 0; r < 8; ++r) {
    float s = ss[r];
    s += __shfl_xor(s, 1);
    s += __shfl_xor(s, 2);
    s += __shfl_xor(s, 4);
    s += __shfl_xor(s, 8);
    if (lm == 0) kn[(size_t)bg * N_ + row0 + r + 8 * half] = s;
  }
}

// ------------------------------------------------------------------
// Kernel 2: attention + softmax + out = adj@h + LN + GELU + residual
// grid (BG, N/16), block 32 (1 wave), dynamic LDS
// ------------------------------------------------------------------
#define SC_OFF   0
#define P16_OFF  (16 * N_ * 4)                 // 65536
#define HST0_OFF (P16_OFF + 16 * N_ * 2)       // 98304
#define HST1_OFF (HST0_OFF + 32 * D_ * 2)      // 114688
#define RS_OFF   (HST1_OFF + 32 * D_ * 2)      // 131072
#define SMEM_SZ  (RS_OFF + 64)                 // 131136

#if USE_ASYNC_LDS
// issue 32 async b128 loads: one 16KB key tile (32 rows x 256 f16) -> LDS
__device__ __forceinline__ void stage_tile_async(const _Float16* src,
                                                 _Float16* dst, int lane) {
#pragma unroll
  for (int j = 0; j < 32; ++j) {
    const _Float16* s = src + ((size_t)(j * 32 + lane)) * 8;  // 16B chunk
    _Float16*       d = dst + ((size_t)(j * 32 + lane)) * 8;
    __builtin_amdgcn_global_load_async_to_lds_b128(
        (AS1 v4i_t*)(uintptr_t)s, (AS3 v4i_t*)(uintptr_t)d, 0, 0);
  }
}
#endif

__global__ void k2_attention(const float* __restrict__ x,
                             const _Float16* __restrict__ h16,
                             const float* __restrict__ kn,
                             const float* __restrict__ lnw,
                             const float* __restrict__ lnb,
                             const float* __restrict__ alpha,
                             const float* __restrict__ lamda,
                             float* __restrict__ out) {
  extern __shared__ char smem[];
  float*    sc   = (float*)(smem + SC_OFF);      // [16][1024] scores; later obuf
  _Float16* p16  = (_Float16*)(smem + P16_OFF);  // [16][1024] probs (f16)
  _Float16* hst0 = (_Float16*)(smem + HST0_OFF); // [32][256] key tile buf 0
  _Float16* hst1 = (_Float16*)(smem + HST1_OFF); // [32][256] key tile buf 1
  float*    rs   = (float*)(smem + RS_OFF);      // [16] 1/rowsum

  const int lane = threadIdx.x & 31;
  const int half = lane >> 4;
  const int lm   = lane & 15;
  const int bg   = blockIdx.x;
  const int b    = bg / G_;
  const int row0 = blockIdx.y * 16;

  const _Float16* hp = h16 + (size_t)bg * N_ * D_;
  const _Float16* qp = h16 + (size_t)(b * G_) * N_ * D_;  // g = 0
  const float*    knp = kn + (size_t)bg * N_;
  const float*    qnp = kn + (size_t)(b * G_) * N_;
  const float     lam = lamda[0];

  // preload q A-fragments for all 8 K-tiles (row = row0+lm)
  v16h aq[8];
  const _Float16* qrow = qp + (size_t)(row0 + lm) * D_;
#pragma unroll
  for (int kt = 0; kt < 8; ++kt) {
#pragma unroll
    for (int vv = 0; vv < 8; ++vv) {
      const h2 t = *(const h2*)(qrow + kt * 32 + a_k0(vv, half));
      aq[kt][2 * vv]     = t.x;
      aq[kt][2 * vv + 1] = t.y;
    }
  }
  float qnv[8];
#pragma unroll
  for (int r = 0; r < 8; ++r) qnv[r] = qnp[row0 + r + 8 * half];

  // ---------- phase 1: scores = -dist/T + lam*I ----------
  const v8f vzero = {0.f, 0.f, 0.f, 0.f, 0.f, 0.f, 0.f, 0.f};
  for (int mt = 0; mt < 64; ++mt) {
    const int tok = mt * 16 + lm;                 // key token for this lane
    const _Float16* hrow = hp + (size_t)tok * D_;
    v16h bf[8];
#pragma unroll
    for (int kt = 0; kt < 8; ++kt) {
#pragma unroll
      for (int vv = 0; vv < 8; ++vv) {
        const h2 t = *(const h2*)(hrow + kt * 32 + b_k0(vv, half));
        bf[kt][2 * vv]     = t.x;
        bf[kt][2 * vv + 1] = t.y;
      }
    }
    v8f c = vzero;
#pragma unroll
    for (int kt = 0; kt < 8; ++kt) c = wmma_f16(aq[kt], bf[kt], c);

    const float knm = knp[tok];
#pragma unroll
    for (int r = 0; r < 8; ++r) {
      const int rloc = r + 8 * half;
      const int rowg = row0 + rloc;
      float sq = qnv[r] + knm - 2.0f * c[r];
      sq = fmaxf(sq, EPS_SQ);
      float s = -sqrtf(sq) * TEMP_INV;
      if (rowg == tok) s += lam;
      sc[rloc * N_ + tok] = s;
    }
  }
  __syncthreads();

  // ---------- phase 2: softmax (each lane: row lm, one half of columns) ----
  {
    const int base = lm * N_ + half * 512;
    float mx = -3.0e38f;
    for (int j = 0; j < 512; ++j) mx = fmaxf(mx, sc[base + j]);
    mx = fmaxf(mx, __shfl_xor(mx, 16));
    float sum = 0.f;
    for (int j = 0; j < 512; ++j) {
      const float p = __expf(sc[base + j] - mx);
      p16[base + j] = (_Float16)p;
      sum += p;
    }
    sum += __shfl_xor(sum, 16);
    if (half == 0) rs[lm] = 1.0f / sum;
  }
  __syncthreads();

  // ---------- phase 3: out = P @ h (two passes of 8 d-tiles) ----------
  float* obuf = sc;                       // reuse dead score buffer: [16][256]
  _Float16* bufs[2] = {hst0, hst1};

  for (int pass = 0; pass < 2; ++pass) {
    const int dbase = pass * 128 + lm;    // runtime part of LDS B address
    v8f od[8];
#pragma unroll
    for (int i = 0; i < 8; ++i) od[i] = vzero;

#if USE_ASYNC_LDS
    // prime the pipeline: tile 0 -> buf 0
    stage_tile_async(hp, bufs[0], lane);
#endif

    for (int kt = 0; kt < 32; ++kt) {
      _Float16* cur = bufs[kt & 1];
#if USE_ASYNC_LDS
      // stream next tile into the other buffer while we compute this one
      if (kt < 31) {
        stage_tile_async(hp + (size_t)(kt + 1) * 32 * D_, bufs[(kt + 1) & 1],
                         lane);
        // async loads complete in order: <=32 outstanding means tile kt landed
        __builtin_amdgcn_s_wait_asynccnt(32);
      } else {
        __builtin_amdgcn_s_wait_asynccnt(0);
      }
#else
      {
        const uint4* src = (const uint4*)(hp + (size_t)kt * 32 * D_);
        uint4* dst = (uint4*)cur;
        if (kt < 31)
          __builtin_prefetch(hp + (size_t)(kt + 1) * 32 * D_ + lane * 16, 0, 0);
#pragma unroll
        for (int j = 0; j < 32; ++j) dst[lane * 32 + j] = src[lane * 32 + j];
      }
      __syncthreads();
#endif

      // A fragment from probabilities (row = lm, K = key index)
      v16h ap;
#pragma unroll
      for (int vv = 0; vv < 8; ++vv) {
        const h2 t = *(const h2*)(p16 + lm * N_ + kt * 32 + a_k0(vv, half));
        ap[2 * vv]     = t.x;
        ap[2 * vv + 1] = t.y;
      }
      // B fragments from LDS: hoisted base, immediate offsets, one burst
      const _Float16* hb = cur + dbase;
      v16h bf[8];
#pragma unroll
      for (int dt = 0; dt < 8; ++dt) {
#pragma unroll
        for (int vv = 0; vv < 8; ++vv) {
          const int K0 = b_k0(vv, half);
          bf[dt][2 * vv]     = hb[(K0)     * D_ + dt * 16];
          bf[dt][2 * vv + 1] = hb[(K0 + 1) * D_ + dt * 16];
        }
      }
#pragma unroll
      for (int dt = 0; dt < 8; ++dt) od[dt] = wmma_f16(ap, bf[dt], od[dt]);
#if !USE_ASYNC_LDS
      __syncthreads();
#endif
    }

    // scale by 1/rowsum, park this half of the columns in LDS
#pragma unroll
    for (int r = 0; r < 8; ++r) {
      const int rloc = r + 8 * half;
      const float rinv = rs[rloc];
#pragma unroll
      for (int dt = 0; dt < 8; ++dt) {
        const int d = pass * 128 + dt * 16 + lm;
        obuf[rloc * D_ + d] = od[dt][r] * rinv;
      }
    }
    __syncthreads();
  }

  // ---------- epilogue: LayerNorm + GELU + residual, straight from LDS ----
  float a = alpha[0];
  a = fminf(fmaxf(a, 0.1f), 0.9f);
  const float one_minus_a = 1.0f - a;

  const float* xg = x + (size_t)bg * N_ * D_;
  float* og = out + (size_t)bg * N_ * D_;

  // lane handles row lm, columns [half*128, half*128+128)
  {
    const int row  = lm;
    const int rowg = row0 + row;
    const int cb   = half * 128;
    const float* orow = obuf + row * D_;

    float s1 = 0.f, s2 = 0.f;
    for (int j = 0; j < 128; ++j) {
      const float o = orow[cb + j];
      s1 += o;
      s2 += o * o;
    }
    s1 += __shfl_xor(s1, 16);
    s2 += __shfl_xor(s2, 16);

    const float mean = s1 * (1.0f / D_);
    float var = s2 * (1.0f / D_) - mean * mean;
    var = fmaxf(var, 0.0f);
    const float rstd = rsqrtf(var + LN_EPS);

    for (int j = 0; j < 128; ++j) {
      const int d = cb + j;
      const float o = orow[d];
      const float v = (o - mean) * rstd * lnw[d] + lnb[d];
      const float g = 0.5f * v * (1.0f + erff(v * 0.70710678118654752f));
      const float xin = xg[(size_t)rowg * D_ + d];
      og[(size_t)rowg * D_ + d] = one_minus_a * xin + a * g;
    }
  }
}

// ------------------------------------------------------------------
extern "C" void kernel_launch(void* const* d_in, const int* in_sizes, int n_in,
                              void* d_out, int out_size, void* d_ws, size_t ws_size,
                              hipStream_t stream) {
  const float* x     = (const float*)d_in[0];
  const float* W     = (const float*)d_in[1];
  const float* bias  = (const float*)d_in[2];
  const float* lnw   = (const float*)d_in[3];
  const float* lnb   = (const float*)d_in[4];
  const float* alpha = (const float*)d_in[5];
  const float* lamda = (const float*)d_in[6];
  float* out = (float*)d_out;

  // workspace partition
  char* ws = (char*)d_ws;
  _Float16* WT  = (_Float16*)(ws);                       // 128 KB
  _Float16* h16 = (_Float16*)(ws + 131072);              // 16 MB
  float*    kn  = (float*)(ws + 131072 + 16777216);      // 128 KB

  k0_transpose_w<<<dim3(D_ * D_ / 256), dim3(256), 0, stream>>>(W, WT);

  k1_gemm_h<<<dim3(BG_, N_ / 64), dim3(128), 0, stream>>>(x, WT, bias, h16, kn);

  k2_attention<<<dim3(BG_, N_ / 16), dim3(32), SMEM_SZ, stream>>>(
      x, h16, kn, lnw, lnb, alpha, lamda, out);
}
